// QRNN_4166118277510
// MI455X (gfx1250) — matrix operand order repaired and checked by reading, not compiled, not run on hardware
//
#include <hip/hip_runtime.h>

// ---------------------------------------------------------------------------
// Fused QRNN (kernel_size=2 conv gates + fo-pooling) for MI455X (gfx1250).
//   pass 1: x (f32) -> bf16 copy in workspace (streaming, one-shot)
//   pass 2: fused gates GEMM (bf16 WMMA, f32 acc) + in-register fo-pool scan
// One workgroup owns (batch b, 64-channel chunk). W-slice is resident in LDS
// (bf16, column-major) for the whole sequence. Time is processed in 64-step
// GEMM stages (4 x 16-row WMMA sub-tiles share every B fragment load 4x).
// A tiles are staged via CDNA5 async global->LDS copies when available.
// ---------------------------------------------------------------------------

typedef __bf16        v16bf __attribute__((ext_vector_type(16)));
typedef float         v8f   __attribute__((ext_vector_type(8)));
typedef unsigned int  v4u   __attribute__((ext_vector_type(4)));
typedef int           v4i   __attribute__((ext_vector_type(4)));

constexpr int B  = 32;
constexpr int S  = 2048;
constexpr int E  = 256;
constexpr int H  = 256;
constexpr int K2 = 2 * E;          // 512  : GEMM K (concat of x_t, x_{t-1})
constexpr int N3 = 3 * H;          // 768  : W column count
constexpr int HC = 64;             // channels per workgroup
constexpr int NCHUNK = H / HC;     // 4 chunks -> 128 workgroups
constexpr int NW  = HC / 16;       // 4 waves per block
constexpr int BLK = 32 * NW;       // 128 threads
constexpr int NCOLS = 3 * HC;      // 192 packed W columns (z|f|o)
constexpr int LDW = K2 + 8;        // padded K-stride (bf16 elems) for W slice
constexpr int LDA = K2 + 8;        // padded K-stride for A tile
constexpr int MT  = 64;            // timesteps per GEMM stage
constexpr int NMT = MT / 16;       // 4 WMMA sub-tiles per stage

#if defined(__has_builtin)
#if __has_builtin(__builtin_amdgcn_global_load_async_to_lds_b128)
#define HAVE_ASYNC_LDS 1
#endif
#endif

__device__ __forceinline__ unsigned short f2bf(float x) {
  union { float f; unsigned u; } v; v.f = x;
  unsigned r = v.u + 0x7FFFu + ((v.u >> 16) & 1u);   // round-to-nearest-even
  return (unsigned short)(r >> 16);
}

__device__ __forceinline__ float sigmoidf_(float x) {
  return 1.0f / (1.0f + __expf(-x));
}

// 16B global -> LDS copy (async on CDNA5; tracked by ASYNCcnt)
__device__ __forceinline__ void copy16_g2l(const void* g, void* l) {
#ifdef HAVE_ASYNC_LDS
  typedef __attribute__((address_space(1))) v4i* gp_t;   // printed: v4i __device__*
  typedef __attribute__((address_space(3))) v4i* lp_t;   // printed: v4i __shared__*
  __builtin_amdgcn_global_load_async_to_lds_b128((gp_t)g, (lp_t)l, 0, 0);
#else
  *(uint4*)l = *(const uint4*)g;
#endif
}

__device__ __forceinline__ void wait_g2l() {
#ifdef HAVE_ASYNC_LDS
#if __has_builtin(__builtin_amdgcn_s_wait_asynccnt)
  __builtin_amdgcn_s_wait_asynccnt(0);
#else
  asm volatile("s_wait_asynccnt 0x0" ::: "memory");
#endif
#endif
}

union Frag { v4u u[2]; v16bf v; };

// ---- pass 1: f32 -> bf16 streaming conversion of x ------------------------
__global__ __launch_bounds__(256)
void cvt_x_bf16(const float* __restrict__ x, unsigned short* __restrict__ xbf,
                int n8) {
  const int i = blockIdx.x * 256 + threadIdx.x;    // 8-element chunk
  if (i >= n8) return;
  const float4 a = ((const float4*)x)[2 * i];
  const float4 b = ((const float4*)x)[2 * i + 1];
  uint4 p;
  p.x = (unsigned)f2bf(a.x) | ((unsigned)f2bf(a.y) << 16);
  p.y = (unsigned)f2bf(a.z) | ((unsigned)f2bf(a.w) << 16);
  p.z = (unsigned)f2bf(b.x) | ((unsigned)f2bf(b.y) << 16);
  p.w = (unsigned)f2bf(b.z) | ((unsigned)f2bf(b.w) << 16);
  ((uint4*)xbf)[i] = p;
}

// ---- pass 2: fused GEMM + scan --------------------------------------------
// PRECVT=true : A staged from bf16 workspace via async global->LDS copies.
// PRECVT=false: A staged from f32 x with in-kernel conversion (fallback).
template <bool PRECVT>
__global__ __launch_bounds__(BLK, 1)
void qrnn_fused(const float* __restrict__ x,
                const unsigned short* __restrict__ xbf,
                const int* __restrict__ mask,
                const float* __restrict__ W, const float* __restrict__ bias,
                float* __restrict__ out) {
  // 192*520*2 = 199,680 B  +  64*520*2 = 66,560 B  = 266,240 B (< 320 KB)
  __shared__ unsigned short Wlds[NCOLS * LDW];
  __shared__ unsigned short Alds[MT * LDA];

  const int tid  = threadIdx.x;
  const int lane = tid & 31;
  const int w    = tid >> 5;        // wave id 0..3
  const int half = lane >> 4;       // lane half (0: lanes 0-15, 1: 16-31)
  const int n    = lane & 15;       // column/channel-within-tile, also A row

  const int b     = blockIdx.x / NCHUNK;
  const int hbase = (blockIdx.x % NCHUNK) * HC;
  const int ch    = hbase + w * 16 + n;            // absolute channel

  float* out_ht = out;                              // [B,S,H]
  float* out_h  = out + (size_t)B * S * H;          // [B,1,H]
  float* out_c  = out_h + (size_t)B * H;            // [B,1,H]

  // ---- Stage this block's W slice into LDS, bf16, column-major ------------
  // packed col c: [0,HC)=z cols, [HC,2HC)=f cols, [2HC,3HC)=o cols
  for (int idx = tid; idx < K2 * (NCOLS / 4); idx += BLK) {
    const int k = idx / (NCOLS / 4);
    const int c = (idx % (NCOLS / 4)) * 4;
    const int gc = (c < HC)     ? (hbase + c)
                 : (c < 2 * HC) ? (H + hbase + (c - HC))
                                : (2 * H + hbase + (c - 2 * HC));
    const float4 v = *(const float4*)(W + (size_t)k * N3 + gc);
    Wlds[(c + 0) * LDW + k] = f2bf(v.x);
    Wlds[(c + 1) * LDW + k] = f2bf(v.y);
    Wlds[(c + 2) * LDW + k] = f2bf(v.z);
    Wlds[(c + 3) * LDW + k] = f2bf(v.w);
  }

  // per-lane gate biases
  const float bz = bias[ch];
  const float bf = bias[H + ch];
  const float bo = bias[2 * H + ch];

  int mi = mask[b] - 1;                     // reference clamps to [0, S-1]
  mi = mi < 0 ? 0 : (mi > S - 1 ? S - 1 : mi);

  const float*          xb  = x   + (size_t)b * S * E;
  const unsigned short* xbb = PRECVT ? (xbf + (size_t)b * S * E) : nullptr;
  float c_in = 0.0f;                        // recurrence carry (all lanes)

  __syncthreads();

  for (int s0 = 0; s0 < S; s0 += MT) {
    // ---- Stage A tile: row r = time s0+r; cols [0,E)=x_t, [E,2E)=x_{t-1} --
    if (PRECVT) {
      // async bf16 copy: 16 B (8 elems) per lane per op, no VGPR round trip
      for (int idx = tid; idx < MT * (K2 / 8); idx += BLK) {
        const int r  = idx >> 6;
        const int c  = (idx & 63) * 8;
        const int t  = (c < E) ? (s0 + r) : (s0 + r - 1);
        const int cc = (c < E) ? c : (c - E);
        if (t >= 0) {
          copy16_g2l(xbb + (size_t)t * E + cc, &Alds[r * LDA + c]);
          if (t + MT < S)                       // warm caches for next stage
            __builtin_prefetch(xbb + (size_t)(t + MT) * E + cc, 0, 3);
        }
      }
      wait_g2l();
      if (s0 == 0 && tid < E / 8) {            // zero x_{t-1} strip of row 0
        *(uint4*)&Alds[E + tid * 8] = make_uint4(0u, 0u, 0u, 0u);
      }
    } else {
      for (int idx = tid; idx < MT * (K2 / 4); idx += BLK) {
        const int r  = idx >> 7;
        const int c  = (idx & 127) * 4;
        const int t  = (c < E) ? (s0 + r) : (s0 + r - 1);
        const int cc = (c < E) ? c : (c - E);
        float4 v = make_float4(0.f, 0.f, 0.f, 0.f);
        if (t >= 0) {
          v = *(const float4*)(xb + (size_t)t * E + cc);
          if (t + MT < S)
            __builtin_prefetch(xb + (size_t)(t + MT) * E + cc, 0, 3);
        }
        unsigned short* dst = &Alds[r * LDA + c];
        const unsigned p0 = (unsigned)f2bf(v.x) | ((unsigned)f2bf(v.y) << 16);
        const unsigned p1 = (unsigned)f2bf(v.z) | ((unsigned)f2bf(v.w) << 16);
        *(uint2*)dst = make_uint2(p0, p1);
      }
    }
    __syncthreads();

    // ---- WMMA: 4 M sub-tiles x 3 chains (z,f,o); each B fragment reused 4x
    v8f acc[NMT][3];
#pragma unroll
    for (int m = 0; m < NMT; ++m) {
      acc[m][0] = v8f{}; acc[m][1] = v8f{}; acc[m][2] = v8f{};
    }
    const unsigned short* bzc = &Wlds[(         w * 16 + n) * LDW];
    const unsigned short* bfc = &Wlds[(HC     + w * 16 + n) * LDW];
    const unsigned short* boc = &Wlds[(2 * HC + w * 16 + n) * LDW];

#pragma unroll
    for (int k = 0; k < K2 / 32; ++k) {
      // B k order used by this code (same as A): lane half h holds contiguous K = 32k+16h .. +16
      Frag Bz, Bf, Bo;
      const unsigned short* zp = bzc + 32 * k + 16 * half;
      Bz.u[0] = *(const v4u*)(zp);  Bz.u[1] = *(const v4u*)(zp + 8);
      const unsigned short* fp = bfc + 32 * k + 16 * half;
      Bf.u[0] = *(const v4u*)(fp);  Bf.u[1] = *(const v4u*)(fp + 8);
      const unsigned short* op = boc + 32 * k + 16 * half;
      Bo.u[0] = *(const v4u*)(op);  Bo.u[1] = *(const v4u*)(op + 8);
#pragma unroll
      for (int m = 0; m < NMT; ++m) {
        // A k order used by this code (same as B): lane half h holds contiguous K = 32k+16h .. +16
        Frag A;
        const unsigned short* ap = &Alds[(16 * m + n) * LDA] + 32 * k + 16 * half;
        A.u[0] = *(const v4u*)(ap);
        A.u[1] = *(const v4u*)(ap + 8);
        acc[m][0] = __builtin_amdgcn_wmma_f32_16x16x32_bf16(
            false, A.v, false, Bz.v, (short)0, acc[m][0], false, false);
        acc[m][1] = __builtin_amdgcn_wmma_f32_16x16x32_bf16(
            false, A.v, false, Bf.v, (short)0, acc[m][1], false, false);
        acc[m][2] = __builtin_amdgcn_wmma_f32_16x16x32_bf16(
            false, A.v, false, Bo.v, (short)0, acc[m][2], false, false);
      }
    }

    // ---- Activations + in-register fo-pooling scan, sub-tile by sub-tile --
    // C/D layout: VGPR g, lane half h -> time = g + 8h, channel = n.
#pragma unroll
    for (int mt = 0; mt < NMT; ++mt) {
      const int sbase = s0 + 16 * mt;
      float z[8], f[8], o[8], cl[8], P[8];
#pragma unroll
      for (int g = 0; g < 8; ++g) {
        z[g] = tanhf(acc[mt][0][g] + bz);
        f[g] = sigmoidf_(acc[mt][1][g] + bf);
        o[g] = sigmoidf_(acc[mt][2][g] + bo);
      }
      // local scan with c_in = 0, plus cumulative decay product:
      //   c_t = cl[t] + P[t] * c_in_of_half
      float cp = 0.0f, pp = 1.0f;
#pragma unroll
      for (int g = 0; g < 8; ++g) {
        cp = f[g] * cp + (1.0f - f[g]) * z[g];
        pp *= f[g];
        cl[g] = cp; P[g] = pp;
      }
      const float c7    = cl[7] + P[7] * c_in;   // end of half0 (all lanes)
      const float cmid  = __shfl(c7, n, 32);     // broadcast half0 tail
      const float cprev = half ? cmid : c_in;

      const int  mrel = mi - sbase;
      const bool mhit = (mrel >= 0) && (mrel < 16) && ((mrel >> 3) == half);
      const int  gsel = mrel & 7;

      float ct7 = 0.0f;
      const size_t rowbase = ((size_t)b * S + sbase + 8 * half) * H + ch;
#pragma unroll
      for (int g = 0; g < 8; ++g) {
        const float ct = cl[g] + P[g] * cprev;
        const float h  = o[g] * ct;
        out_ht[rowbase + (size_t)g * H] = h;
        if (g == 7) ct7 = ct;
        if (mhit && g == gsel) {
          out_h[(size_t)b * H + ch] = h;
          out_c[(size_t)b * H + ch] = ct;
        }
      }
      c_in = __shfl(ct7, 16 + n, 32);            // t=15 cell -> next carry
    }
    __syncthreads();
  }
}

extern "C" void kernel_launch(void* const* d_in, const int* in_sizes, int n_in,
                              void* d_out, int out_size, void* d_ws, size_t ws_size,
                              hipStream_t stream) {
  (void)in_sizes; (void)n_in; (void)out_size;
  const float* x    = (const float*)d_in[0];
  const int*   mask = (const int*)d_in[1];
  const float* W    = (const float*)d_in[2];
  const float* bias = (const float*)d_in[3];
  float*       out  = (float*)d_out;

  const size_t need = (size_t)B * S * E * sizeof(unsigned short);  // 32 MB
  if (ws_size >= need) {
    unsigned short* xbf = (unsigned short*)d_ws;
    const int n8 = B * S * E / 8;                 // 2,097,152 chunks
    cvt_x_bf16<<<dim3((n8 + 255) / 256), dim3(256), 0, stream>>>(x, xbf, n8);
    qrnn_fused<true><<<dim3(B * NCHUNK), dim3(BLK), 0, stream>>>(
        x, xbf, mask, W, bias, out);
  } else {
    qrnn_fused<false><<<dim3(B * NCHUNK), dim3(BLK), 0, stream>>>(
        x, nullptr, mask, W, bias, out);
  }
}
